// AttentionConv_61753039782642
// MI455X (gfx1250) — matrix operand order repaired and checked by reading, not compiled, not run on hardware
//
#include <hip/hip_runtime.h>

typedef __attribute__((ext_vector_type(16))) _Float16 v16h;
typedef __attribute__((ext_vector_type(8)))  _Float16 v8h;
typedef __attribute__((ext_vector_type(4)))  _Float16 v4h;
typedef __attribute__((ext_vector_type(8)))  float    v8f;

// Types for the async global->LDS builtin: it takes pointers to 16-byte
// GCC-style int vectors, in AS1 (global / __device__) and AS3 (LDS).
typedef int v4i_g __attribute__((vector_size(16)));
typedef __attribute__((address_space(1))) v4i_g as1_v4i;
typedef __attribute__((address_space(3))) v4i_g as3_v4i;

#define NB    4
#define NHEAD 2
#define NPIX  4096          // 64*64 pixels per batch image
#define CIN   128
#define HD    64            // head dim
#define NTOT  (NB * NPIX)   // 16384 total pixels
#define KT    64            // keys staged per block iteration

#if defined(__has_builtin)
#if __has_builtin(__builtin_amdgcn_global_load_async_to_lds_b128) && \
    __has_builtin(__builtin_amdgcn_s_wait_asynccnt)
#define USE_ASYNC 1
#endif
#endif

// ---------------------------------------------------------------------------
// WMMA fragment loaders (wave32, v_wmma_f32_16x16x32_f16)
// A (16Mx32K f16): lanes 0-15 rows 0-15 w/ K 0..7 (v0-3), 16..23 (v4-7);
//                  lanes 16-31 same rows w/ K 8..15 / 24..31.
// B (32Kx16N f16): lane owns column (lane&15); lanes 0-15 K 0..7 / 16..23,
//                  lanes 16-31 K 8..15 / 24..31 (same as A).
// C/D (16x16 f32): VGPR r, lanes 0-15 -> row r; lanes 16-31 -> row r+8.
// ---------------------------------------------------------------------------
static __device__ __forceinline__ v16h ldA_h(const _Float16* tile, int ld) {
  const int l = threadIdx.x & 31;
  const _Float16* p = tile + (l & 15) * ld + ((l >> 4) << 3);
  v8h lo = *(const v8h*)(p);
  v8h hi = *(const v8h*)(p + 16);
  v16h r;
#pragma unroll
  for (int i = 0; i < 8; ++i) { r[i] = lo[i]; r[i + 8] = hi[i]; }
  return r;
}

static __device__ __forceinline__ v16h ldB_h(const _Float16* tile, int ld) {
  const int l = threadIdx.x & 31;
  const _Float16* p = tile + (l & 15) * ld + ((l >> 4) << 3);
  v8h lo = *(const v8h*)(p);
  v8h hi = *(const v8h*)(p + 16);
  v16h r;
#pragma unroll
  for (int i = 0; i < 8; ++i) { r[i] = lo[i]; r[i + 8] = hi[i]; }
  return r;
}

static __device__ __forceinline__ v8f wmma16(v16h a, v16h b, v8f c) {
  return __builtin_amdgcn_wmma_f32_16x16x32_f16(false, a, false, b,
                                                (short)0, c, false, false);
}

// 16-byte async global->LDS copy (falls back to a synchronous copy).
static __device__ __forceinline__ void async_cp16(const _Float16* g,
                                                  _Float16* l) {
#ifdef USE_ASYNC
  __builtin_amdgcn_global_load_async_to_lds_b128(
      (as1_v4i*)(uintptr_t)g,
      (as3_v4i*)(unsigned)(uintptr_t)l, 0, 0);
#else
  *(v8h*)l = *(const v8h*)g;
#endif
}

template <int N>
static __device__ __forceinline__ void wait_async() {
#ifdef USE_ASYNC
  __builtin_amdgcn_s_wait_asynccnt(N);
#endif
}

// ---------------------------------------------------------------------------
// Kernel 0a: x f32 -> f16 (done once; qkv_proj re-reads each tile 24x)
// ---------------------------------------------------------------------------
__global__ void cvt_x(const float* __restrict__ x, _Float16* __restrict__ xh) {
  const int i = (blockIdx.x * 256 + threadIdx.x) * 4;
  const float4 v = *(const float4*)(x + i);
  v4h o;
  o[0] = (_Float16)v.x; o[1] = (_Float16)v.y;
  o[2] = (_Float16)v.z; o[3] = (_Float16)v.w;
  *(v4h*)(xh + i) = o;
}

// ---------------------------------------------------------------------------
// Kernel 0b: weight prep — transpose + convert to f16.
// ---------------------------------------------------------------------------
__global__ void prep_w(const float* __restrict__ wqkv,
                       const float* __restrict__ wout,
                       _Float16* __restrict__ wqkvT,
                       _Float16* __restrict__ woutT) {
  int i = blockIdx.x * 256 + threadIdx.x;
  if (i < 128 * 384) {
    int d = i / 128, c = i % 128;
    wqkvT[i] = (_Float16)wqkv[c * 384 + d];
  }
  int j = i - 128 * 384;
  if (j >= 0 && j < 128 * 128) {
    int d = j / 128, c = j % 128;
    woutT[j] = (_Float16)wout[c * 128 + d];
  }
}

// ---------------------------------------------------------------------------
// Kernel 1: QKV projection. One wave per 16x16 tile, 4 K-steps of 32.
// Q is pre-scaled by C^-0.5 * log2(e) so softmax runs in exp2 domain.
// Q,K: [b][h][n][64] f16 row-major; V: transposed [b][h][64][n] f16.
// ---------------------------------------------------------------------------
__global__ __launch_bounds__(256) void qkv_proj(
    const _Float16* __restrict__ xh, const _Float16* __restrict__ wT,
    _Float16* __restrict__ Q, _Float16* __restrict__ K,
    _Float16* __restrict__ Vt) {
  const int wave = threadIdx.x >> 5;
  const int tile = blockIdx.x * 8 + wave;
  const int mt = tile / 24, nt = tile % 24;
  const int m0 = mt * 16, d0 = nt * 16;
  const int l = threadIdx.x & 31;

  v8f acc = {};
#pragma unroll
  for (int kc = 0; kc < 4; ++kc) {
    v16h a = ldA_h(xh + (size_t)m0 * CIN + kc * 32, CIN);
    v16h b = ldB_h(wT + (size_t)d0 * CIN + kc * 32, CIN);
    acc = wmma16(a, b, acc);
  }

  const int head = d0 / 192;
  const int rr   = d0 % 192;
  const int kind = rr / 64;                     // 0=q 1=k 2=v
  const int col  = (rr % 64) + (l & 15);
  const int rowbase = m0 + ((l >> 4) << 3);
  const float qscale = 0.08838834764831845f * 1.4426950408889634f;

  if (kind == 2) {
    const int b = rowbase >> 12, n = rowbase & (NPIX - 1);
    v8h pk;
#pragma unroll
    for (int i = 0; i < 8; ++i) pk[i] = (_Float16)acc[i];
    *(v8h*)(Vt + ((size_t)((b * NHEAD + head) * HD + col)) * NPIX + n) = pk;
  } else {
    _Float16* dst = (kind == 0) ? Q : K;
    const float s = (kind == 0) ? qscale : 1.0f;
#pragma unroll
    for (int i = 0; i < 8; ++i) {
      const int pix = rowbase + i;
      const int b = pix >> 12, n = pix & (NPIX - 1);
      dst[((size_t)(b * NHEAD + head) * NPIX + n) * HD + col] =
          (_Float16)(acc[i] * s);
    }
  }
}

// ---------------------------------------------------------------------------
// Kernel 2: fused flash attention per (batch, head).
// Block = 8 waves = 128 query rows; keys streamed in 64-wide tiles staged
// block-cooperatively into LDS with double-buffered async global->LDS copies.
// Per wave per tile: 8 WMMA (S) + 8 WMMA (O), online softmax in exp2 domain.
// ---------------------------------------------------------------------------
__global__ __launch_bounds__(256) void flash_attn(
    const _Float16* __restrict__ Q, const _Float16* __restrict__ K,
    const _Float16* __restrict__ Vt, _Float16* __restrict__ res) {
  __shared__ __align__(16) _Float16 LdsK[2][KT * HD];   // [key][feat]
  __shared__ __align__(16) _Float16 LdsV[2][HD * KT];   // [feat][key]
  __shared__ __align__(16) _Float16 Plds[8][16 * KT];   // per-wave P slab

  const int wave = threadIdx.x >> 5;
  const int l = threadIdx.x & 31;
  const int b = blockIdx.z, h = blockIdx.y;
  const int q0 = blockIdx.x * 128 + wave * 16;

  const _Float16* Qb = Q  + (size_t)(b * NHEAD + h) * NPIX * HD;
  const _Float16* Kb = K  + (size_t)(b * NHEAD + h) * NPIX * HD;
  const _Float16* Vb = Vt + (size_t)(b * NHEAD + h) * HD * NPIX;

  // Stage one K/V tile: 64x64 f16 each = 8KB+8KB, 4 b128 chunks per thread.
  auto stage = [&](int buf, int j0) {
    const int t = threadIdx.x;
#pragma unroll
    for (int i = 0; i < 2; ++i) {
      const int c = t + 256 * i;
      const int row = c >> 3, off = (c & 7) << 3;
      async_cp16(Kb + (size_t)(j0 + row) * HD + off, &LdsK[buf][row * HD + off]);
      async_cp16(Vb + (size_t)row * NPIX + j0 + off, &LdsV[buf][row * HD + off]);
    }
  };

  // Resident Q tile A-fragments (16x64 = two K-chunks)
  const v16h qa0 = ldA_h(Qb + (size_t)q0 * HD, HD);
  const v16h qa1 = ldA_h(Qb + (size_t)q0 * HD + 32, HD);

  float mrun[8], lrun[8];
  v8f acc[4];
#pragma unroll
  for (int i = 0; i < 8; ++i) { mrun[i] = -1e30f; lrun[i] = 0.0f; }
#pragma unroll
  for (int f = 0; f < 4; ++f) acc[f] = (v8f){};

  _Float16* P = Plds[wave];
  const int NT = NPIX / KT;       // 64 iterations

  stage(0, 0);
#pragma unroll 2
  for (int jt = 0; jt < NT; ++jt) {
    const int cur = jt & 1;
    if (jt + 1 < NT) {
      stage(cur ^ 1, (jt + 1) * KT);
      wait_async<4>();            // tile `cur` complete (async loads in-order)
    } else {
      wait_async<0>();
    }
    __syncthreads();              // staging visible block-wide

    const _Float16* ldsK = LdsK[cur];
    const _Float16* ldsV = LdsV[cur];

    // S = Q(16x64) @ K^T(64x64): four 16x16 D tiles, two K-chunks each
    v8f s[4];
#pragma unroll
    for (int t = 0; t < 4; ++t) {
      v8f z = {};
      z = wmma16(qa0, ldB_h(ldsK + t * 16 * HD, HD), z);
      z = wmma16(qa1, ldB_h(ldsK + t * 16 * HD + 32, HD), z);
      s[t] = z;
    }

    // Online softmax (exp2 domain); row r lives across one 16-lane half.
#pragma unroll
    for (int r = 0; r < 8; ++r) {
      float mx = fmaxf(fmaxf(s[0][r], s[1][r]), fmaxf(s[2][r], s[3][r]));
#pragma unroll
      for (int msk = 8; msk >= 1; msk >>= 1)
        mx = fmaxf(mx, __shfl_xor(mx, msk, 32));
      const float mn   = fmaxf(mrun[r], mx);
      const float corr = exp2f(mrun[r] - mn);
      float p[4], rs = 0.0f;
#pragma unroll
      for (int t = 0; t < 4; ++t) { p[t] = exp2f(s[t][r] - mn); rs += p[t]; }
#pragma unroll
      for (int msk = 8; msk >= 1; msk >>= 1)
        rs += __shfl_xor(rs, msk, 32);
      lrun[r] = lrun[r] * corr + rs;
      mrun[r] = mn;
      const int row = r + ((l >> 4) << 3);
#pragma unroll
      for (int t = 0; t < 4; ++t)
        P[row * KT + (l & 15) + t * 16] = (_Float16)p[t];
#pragma unroll
      for (int f = 0; f < 4; ++f) acc[f][r] *= corr;
    }

    // O += P(16x64) @ V(64x64): P re-read in A-layout from the LDS slab.
    const v16h pa0 = ldA_h(P, KT);
    const v16h pa1 = ldA_h(P + 32, KT);
#pragma unroll
    for (int f = 0; f < 4; ++f) {
      acc[f] = wmma16(pa0, ldB_h(ldsV + f * 16 * HD, HD), acc[f]);
      acc[f] = wmma16(pa1, ldB_h(ldsV + f * 16 * HD + 32, HD), acc[f]);
    }

    __syncthreads();              // done reading `cur` before it is restaged
  }

  // Normalize and emit res[b*NPIX + row][h*64 + f*16 + col] as f16.
#pragma unroll
  for (int f = 0; f < 4; ++f) {
#pragma unroll
    for (int r = 0; r < 8; ++r) {
      const int row = q0 + r + ((l >> 4) << 3);
      const float o = acc[f][r] / lrun[r];
      res[((size_t)(b * NPIX + row)) * CIN + h * HD + f * 16 + (l & 15)] =
          (_Float16)o;
    }
  }
}

// ---------------------------------------------------------------------------
// Kernel 3: output projection + bias, f32 out.
// ---------------------------------------------------------------------------
__global__ __launch_bounds__(256) void out_proj(
    const _Float16* __restrict__ res, const _Float16* __restrict__ wT,
    const float* __restrict__ bias, float* __restrict__ out) {
  const int wave = threadIdx.x >> 5;
  const int tile = blockIdx.x * 8 + wave;
  const int mt = tile >> 3, nt = tile & 7;
  const int m0 = mt * 16, d0 = nt * 16;
  const int l = threadIdx.x & 31;

  v8f acc = {};
#pragma unroll
  for (int kc = 0; kc < 4; ++kc) {
    v16h a = ldA_h(res + (size_t)m0 * CIN + kc * 32, CIN);
    v16h b = ldB_h(wT + (size_t)d0 * CIN + kc * 32, CIN);
    acc = wmma16(a, b, acc);
  }

  const float bb = bias[d0 + (l & 15)];
  const int rowbase = m0 + ((l >> 4) << 3);
#pragma unroll
  for (int i = 0; i < 8; ++i)
    out[(size_t)(rowbase + i) * CIN + d0 + (l & 15)] = acc[i] + bb;
}

// ---------------------------------------------------------------------------
// Launcher
// ---------------------------------------------------------------------------
extern "C" void kernel_launch(void* const* d_in, const int* in_sizes, int n_in,
                              void* d_out, int out_size, void* d_ws,
                              size_t ws_size, hipStream_t stream) {
  const float* x    = (const float*)d_in[0];   // [4,64,64,128]
  const float* wqkv = (const float*)d_in[1];   // [128,384]
  const float* wout = (const float*)d_in[2];   // [128,128]
  const float* bout = (const float*)d_in[3];   // [128]
  float* out = (float*)d_out;                  // [4,64,64,128] f32

  _Float16* ws = (_Float16*)d_ws;
  const size_t QSZ = (size_t)NB * NHEAD * NPIX * HD;   // 2,097,152 halves
  _Float16* Q     = ws;
  _Float16* K     = Q + QSZ;
  _Float16* Vt    = K + QSZ;
  _Float16* res   = Vt + QSZ;
  _Float16* wqkvT = res + (size_t)NTOT * CIN;
  _Float16* woutT = wqkvT + (size_t)384 * 128;
  _Float16* xh    = woutT + (size_t)128 * 128;

  cvt_x<<<(NTOT * CIN) / (256 * 4), 256, 0, stream>>>(x, xh);
  prep_w<<<256, 256, 0, stream>>>(wqkv, wout, wqkvT, woutT);

  qkv_proj<<<(NTOT / 16) * 24 / 8, 256, 0, stream>>>(xh, wqkvT, Q, K, Vt);

  dim3 g2(NPIX / 128, NHEAD, NB);
  flash_attn<<<g2, 256, 0, stream>>>(Q, K, Vt, res);

  out_proj<<<(NTOT / 16) * (CIN / 16) / 8, 256, 0, stream>>>(res, woutT, bout,
                                                             out);
}
